// TransformerShell_7206955123256
// MI455X (gfx1250) — matrix operand order repaired and checked by reading, not compiled, not run on hardware
//
#include <hip/hip_runtime.h>
#include <hip/hip_bf16.h>
#include <math.h>

typedef __attribute__((ext_vector_type(16))) _Float16 v16h;
typedef __attribute__((ext_vector_type(8)))  float    v8f;

// ---------------------------------------------------------------------------
// Float atomic max via signed/unsigned int ordering trick (init to -inf).
// ---------------------------------------------------------------------------
__device__ __forceinline__ void atomicMaxF(float* addr, float val) {
    if (val >= 0.0f) atomicMax((int*)addr, __float_as_int(val));
    else             atomicMin((unsigned int*)addr, __float_as_uint(val));
}

__device__ __forceinline__ float waveReduceSum(float v) {
#pragma unroll
    for (int off = 16; off > 0; off >>= 1) v += __shfl_xor(v, off, 32);
    return v;
}

// ---------------------------------------------------------------------------
// Generic WMMA GEMM: out[N,Cout] = in[N,Cin=64] @ W[Cout,Cin]^T + bias[Cout]
// One wave computes one 16-row tile across all Cout/16 column tiles.
// Two K=32 steps of v_wmma_f32_16x16x32_f16 cover Cin=64.
// A fragment layout (16-bit A 16x32, ISA 7.12.2): lane half p=lane>>4,
//   VGPR j packs K = (j>>2)*16 + p*8 + (j&3)*2 (+1).
// B fragment layout (16-bit B 32x16): lane half p, VGPR j packs K = (j>>2)*16 + p*8 + (j&3)*2 (+1).
// D layout: VGPR r -> row p*8+r, col = lane&15.
// ---------------------------------------------------------------------------
__global__ void gemm_wmma_f16(const float* __restrict__ in,
                              const float* __restrict__ W,
                              const float* __restrict__ bias,
                              float* __restrict__ out,
                              int nTiles, int Cin, int Cout) {
    int wave = threadIdx.x >> 5;
    int tile = blockIdx.x * (blockDim.x >> 5) + wave;
    if (tile >= nTiles) return;                 // wave-uniform: EXEC all-1 inside
    int lane = threadIdx.x & 31;
    int p    = lane >> 4;
    int mr   = lane & 15;

    // ---- Load the two A fragments (K-step 0 and 1) for this node tile ----
    const float* xr = in + (size_t)(tile * 16 + mr) * Cin;
    v16h a0, a1;
#pragma unroll
    for (int j = 0; j < 8; ++j) {
        int k = ((j >> 2) << 4) + (p << 3) + ((j & 3) << 1);
        float2 f0 = *(const float2*)(xr + k);
        float2 f1 = *(const float2*)(xr + 32 + k);
        a0[2 * j]     = (_Float16)f0.x;  a0[2 * j + 1] = (_Float16)f0.y;
        a1[2 * j]     = (_Float16)f1.x;  a1[2 * j + 1] = (_Float16)f1.y;
    }

    int nCt = Cout >> 4;
    for (int ct = 0; ct < nCt; ++ct) {
        int n = ct * 16 + mr;
        const float* wr = W + (size_t)n * Cin;
        v16h bf0, bf1;
#pragma unroll
        for (int j = 0; j < 8; ++j) {
            int k = ((j >> 2) << 4) + (p << 3) + ((j & 3) << 1);
            float2 f0 = *(const float2*)(wr + k);
            float2 f1 = *(const float2*)(wr + 32 + k);
            bf0[2 * j]     = (_Float16)f0.x;  bf0[2 * j + 1] = (_Float16)f0.y;
            bf1[2 * j]     = (_Float16)f1.x;  bf1[2 * j + 1] = (_Float16)f1.y;
        }
        v8f c = {};
        c = __builtin_amdgcn_wmma_f32_16x16x32_f16(false, a0, false, bf0,
                                                   (short)0, c, false, false);
        c = __builtin_amdgcn_wmma_f32_16x16x32_f16(false, a1, false, bf1,
                                                   (short)0, c, false, false);
        float bn = bias[n];
#pragma unroll
        for (int r = 0; r < 8; ++r) {
            int mo = tile * 16 + (p << 3) + r;
            out[(size_t)mo * Cout + ct * 16 + mr] = c[r] + bn;
        }
    }
}

// ---------------------------------------------------------------------------
// Init: agg[N*128]=0, amax[N*2]=-inf, denom[N*2]=0
// ---------------------------------------------------------------------------
__global__ void init_ws(float* __restrict__ agg, float* __restrict__ amax,
                        float* __restrict__ denom, int Nn) {
    int t = blockIdx.x * blockDim.x + threadIdx.x;
    if (t < Nn * 128) agg[t] = 0.0f;
    if (t < Nn * 2) { amax[t] = -__builtin_inff(); denom[t] = 0.0f; }
}

// ---------------------------------------------------------------------------
// Edge pass 1: alpha[e,h] = (q[dst,h,:] . (k[src,h,:] + ea*We[h,:]))/8
//              amax[dst,h] = max(...). One wave per edge.
// ---------------------------------------------------------------------------
__global__ void edge_alpha_max(const int* __restrict__ ei,
                               const float* __restrict__ ea,
                               const float* __restrict__ q,
                               const float* __restrict__ kk,
                               const float* __restrict__ We,
                               float* __restrict__ alpha,
                               float* __restrict__ amax, int Ee) {
    int wid = blockIdx.x * (blockDim.x >> 5) + (threadIdx.x >> 5);
    if (wid >= Ee) return;
    int lane = threadIdx.x & 31;
    int s = ei[wid];
    int d = ei[Ee + wid];
    float eav = ea[wid];
    if (wid + 8192 < Ee) {                       // stream edge indices ahead
        __builtin_prefetch(&ei[wid + 8192], 0, 1);
        __builtin_prefetch(&ei[Ee + wid + 8192], 0, 1);
    }
    const float* qd = q  + (size_t)d * 128;
    const float* ks = kk + (size_t)s * 128;
#pragma unroll
    for (int h = 0; h < 2; ++h) {
        float part = 0.0f;
#pragma unroll
        for (int cc = 0; cc < 2; ++cc) {
            int c = h * 64 + lane + cc * 32;
            part += qd[c] * (ks[c] + eav * We[c]);
        }
        part = waveReduceSum(part);
        if (lane == 0) {
            float al = part * 0.125f;            // / sqrt(64)
            alpha[wid * 2 + h] = al;
            atomicMaxF(&amax[d * 2 + h], al);
        }
    }
}

// ---------------------------------------------------------------------------
// Edge pass 2: ex = exp(alpha - amax[dst]); denom[dst] += ex. Thread/(e,h).
// ---------------------------------------------------------------------------
__global__ void edge_exp_sum(const int* __restrict__ ei,
                             float* __restrict__ alpha,
                             const float* __restrict__ amax,
                             float* __restrict__ denom, int Ee) {
    int t = blockIdx.x * blockDim.x + threadIdx.x;
    if (t >= 2 * Ee) return;
    int e = t >> 1, h = t & 1;
    int d = ei[Ee + e];
    float ex = __expf(alpha[t] - amax[d * 2 + h]);
    alpha[t] = ex;
    atomicAdd(&denom[d * 2 + h], ex);
}

// ---------------------------------------------------------------------------
// Edge pass 3: agg[dst,h,:] += (v[src,h,:] + ea*We[h,:]) * ex/denom[dst,h]
// ---------------------------------------------------------------------------
__global__ void edge_scatter(const int* __restrict__ ei,
                             const float* __restrict__ ea,
                             const float* __restrict__ vv,
                             const float* __restrict__ We,
                             const float* __restrict__ alpha,
                             const float* __restrict__ denom,
                             float* __restrict__ agg, int Ee) {
    int wid = blockIdx.x * (blockDim.x >> 5) + (threadIdx.x >> 5);
    if (wid >= Ee) return;
    int lane = threadIdx.x & 31;
    int s = ei[wid];
    int d = ei[Ee + wid];
    float eav = ea[wid];
#pragma unroll
    for (int h = 0; h < 2; ++h) {
        float w = alpha[wid * 2 + h] / denom[d * 2 + h];
#pragma unroll
        for (int cc = 0; cc < 2; ++cc) {
            int c = h * 64 + lane + cc * 32;
            float val = (vv[(size_t)s * 128 + c] + eav * We[c]) * w;
            atomicAdd(&agg[(size_t)d * 128 + c], val);
        }
    }
}

// ---------------------------------------------------------------------------
// Node pass: h = mean_heads(agg) + skip ; x1 = x + LN(h; g0,b0). Wave/node.
// ---------------------------------------------------------------------------
__global__ void node_ln0(const float* __restrict__ x,
                         const float* __restrict__ agg,
                         const float* __restrict__ skip,
                         const float* __restrict__ g0,
                         const float* __restrict__ b0,
                         float* __restrict__ x1, int Nn) {
    int wid = blockIdx.x * (blockDim.x >> 5) + (threadIdx.x >> 5);
    if (wid >= Nn) return;
    int lane = threadIdx.x & 31;
    const float* ag = agg + (size_t)wid * 128;
    float h0 = (ag[lane]      + ag[64 + lane])      * 0.5f + skip[(size_t)wid * 64 + lane];
    float h1 = (ag[lane + 32] + ag[96 + lane])      * 0.5f + skip[(size_t)wid * 64 + lane + 32];
    float mean = waveReduceSum(h0 + h1) * (1.0f / 64.0f);
    float d0 = h0 - mean, d1 = h1 - mean;
    float var = waveReduceSum(d0 * d0 + d1 * d1) * (1.0f / 64.0f);
    float inv = rsqrtf(var + 1e-5f);
    x1[(size_t)wid * 64 + lane]      = x[(size_t)wid * 64 + lane]      + d0 * inv * g0[lane]      + b0[lane];
    x1[(size_t)wid * 64 + lane + 32] = x[(size_t)wid * 64 + lane + 32] + d1 * inv * g0[lane + 32] + b0[lane + 32];
}

// ---------------------------------------------------------------------------
// Final: out = x1 + LN(t; g1,b1)
// ---------------------------------------------------------------------------
__global__ void node_ln1(const float* __restrict__ x1,
                         const float* __restrict__ t,
                         const float* __restrict__ g1,
                         const float* __restrict__ b1,
                         float* __restrict__ out, int Nn) {
    int wid = blockIdx.x * (blockDim.x >> 5) + (threadIdx.x >> 5);
    if (wid >= Nn) return;
    int lane = threadIdx.x & 31;
    float h0 = t[(size_t)wid * 64 + lane];
    float h1 = t[(size_t)wid * 64 + lane + 32];
    float mean = waveReduceSum(h0 + h1) * (1.0f / 64.0f);
    float d0 = h0 - mean, d1 = h1 - mean;
    float var = waveReduceSum(d0 * d0 + d1 * d1) * (1.0f / 64.0f);
    float inv = rsqrtf(var + 1e-5f);
    out[(size_t)wid * 64 + lane]      = x1[(size_t)wid * 64 + lane]      + d0 * inv * g1[lane]      + b1[lane];
    out[(size_t)wid * 64 + lane + 32] = x1[(size_t)wid * 64 + lane + 32] + d1 * inv * g1[lane + 32] + b1[lane + 32];
}

// ---------------------------------------------------------------------------
extern "C" void kernel_launch(void* const* d_in, const int* in_sizes, int n_in,
                              void* d_out, int out_size, void* d_ws, size_t ws_size,
                              hipStream_t stream) {
    const float* x     = (const float*)d_in[0];
    const int*   ei    = (const int*)  d_in[1];   // [2,E]
    const float* ea    = (const float*)d_in[2];   // [E,1]
    const float* Wq    = (const float*)d_in[3];
    const float* bq    = (const float*)d_in[4];
    const float* Wk    = (const float*)d_in[5];
    const float* bk    = (const float*)d_in[6];
    const float* Wv    = (const float*)d_in[7];
    const float* bv    = (const float*)d_in[8];
    const float* We    = (const float*)d_in[9];   // [128,1] -> flat [128]
    const float* Wskip = (const float*)d_in[10];
    const float* bskip = (const float*)d_in[11];
    const float* Wl    = (const float*)d_in[12];
    const float* bl    = (const float*)d_in[13];
    const float* g0    = (const float*)d_in[14];
    const float* b0    = (const float*)d_in[15];
    const float* g1    = (const float*)d_in[16];
    const float* b1    = (const float*)d_in[17];
    float* out = (float*)d_out;

    const int Nn = in_sizes[0] / 64;     // 50000
    const int Ee = in_sizes[2];          // 800000

    // Workspace layout (floats). x1 aliases q, t aliases k (dead after pass 3).
    float* ws = (float*)d_ws;
    size_t off = 0;
    float* qb    = ws + off; off += (size_t)Nn * 128;
    float* kb    = ws + off; off += (size_t)Nn * 128;
    float* vb    = ws + off; off += (size_t)Nn * 128;
    float* skipb = ws + off; off += (size_t)Nn * 64;
    float* aggb  = ws + off; off += (size_t)Nn * 128;
    float* amaxb = ws + off; off += (size_t)Nn * 2;
    float* denb  = ws + off; off += (size_t)Nn * 2;
    float* alph  = ws + off; off += (size_t)Ee * 2;
    float* x1b   = qb;                   // alias
    float* tb    = kb;                   // alias

    const int nTiles = (Nn + 15) / 16;   // 3125 (N divisible by 16)
    dim3 blk(256);
    dim3 gGemm((unsigned)((nTiles + 7) / 8));
    dim3 gEdgeW((unsigned)((Ee + 7) / 8));       // 8 waves / block
    dim3 gEdgeT((unsigned)((2 * Ee + 255) / 256));
    dim3 gNode((unsigned)((Nn + 7) / 8));
    dim3 gInit((unsigned)(((size_t)Nn * 128 + 255) / 256));

    init_ws<<<gInit, blk, 0, stream>>>(aggb, amaxb, denb, Nn);

    gemm_wmma_f16<<<gGemm, blk, 0, stream>>>(x, Wq, bq, qb, nTiles, 64, 128);
    gemm_wmma_f16<<<gGemm, blk, 0, stream>>>(x, Wk, bk, kb, nTiles, 64, 128);
    gemm_wmma_f16<<<gGemm, blk, 0, stream>>>(x, Wv, bv, vb, nTiles, 64, 128);
    gemm_wmma_f16<<<gGemm, blk, 0, stream>>>(x, Wskip, bskip, skipb, nTiles, 64, 64);

    edge_alpha_max<<<gEdgeW, blk, 0, stream>>>(ei, ea, qb, kb, We, alph, amaxb, Ee);
    edge_exp_sum  <<<gEdgeT, blk, 0, stream>>>(ei, alph, amaxb, denb, Ee);
    edge_scatter  <<<gEdgeW, blk, 0, stream>>>(ei, ea, vb, We, alph, denb, aggb, Ee);

    node_ln0<<<gNode, blk, 0, stream>>>(x, aggb, skipb, g0, b0, x1b, Nn);
    gemm_wmma_f16<<<gGemm, blk, 0, stream>>>(x1b, Wl, bl, tb, nTiles, 64, 64);
    node_ln1<<<gNode, blk, 0, stream>>>(x1b, tb, g1, b1, out, Nn);
}
